// Block_89824946029265
// MI455X (gfx1250) — matrix operand order repaired and checked by reading, not compiled, not run on hardware
//
#include <hip/hip_runtime.h>

#define D 1024
#define V 64
#define NTOK (8 * 4096)

typedef __attribute__((ext_vector_type(16))) __bf16        v16bf;
typedef __attribute__((ext_vector_type(8)))  float         v8f;
typedef __attribute__((ext_vector_type(8)))  unsigned int  v8u;
typedef __attribute__((ext_vector_type(4)))  unsigned int  v4u;

static __device__ __forceinline__ unsigned short f2bf(float f) {
    union { float f; unsigned u; } v; v.f = f;
    unsigned r = v.u + 0x7FFFu + ((v.u >> 16) & 1u);   // round-to-nearest-even
    return (unsigned short)(r >> 16);
}

static __device__ __forceinline__ float fast_rcp(float x) {
#if __has_builtin(__builtin_amdgcn_rcpf)
    return __builtin_amdgcn_rcpf(x);     // single v_rcp_f32, no IEEE div expansion
#else
    return 1.0f / x;
#endif
}

static __device__ __forceinline__ float sigmoidf(float z) {
    return fast_rcp(1.0f + __expf(-z));
}

static __device__ __forceinline__ v16bf as16bf(v8u u) {
    return __builtin_bit_cast(v16bf, u);
}

static __device__ __forceinline__ v8f wmma_bf16(v16bf a, v16bf b, v8f c) {
    // (neg_a, A, neg_b, B, c_mod, C, reuse_a, reuse_b)
    return __builtin_amdgcn_wmma_f32_16x16x32_bf16(false, a, false, b, (short)0, c, false, false);
}

// ---------------------------------------------------------------------------
// Prep: head_weight fp32[64][1024] -> bf16 [64][1024]
//       emb_weight  fp32[64][1024] -> bf16 transposed [1024][64]
// ---------------------------------------------------------------------------
__global__ void prep_weights(const float* __restrict__ head_w,
                             const float* __restrict__ emb_w,
                             unsigned short* __restrict__ head_bf,
                             unsigned short* __restrict__ embT_bf) {
    int i = blockIdx.x * blockDim.x + threadIdx.x;   // 65536 total
    if (i < V * D) {
        head_bf[i] = f2bf(head_w[i]);
        int v = i / D, d = i % D;
        embT_bf[d * V + v] = f2bf(emb_w[i]);
    }
}

// ---------------------------------------------------------------------------
// Fused: gate-blend -> logits WMMA -> softmax/argmax -> soft_emb WMMA -> blend
// 256 threads = 8 waves; wave owns 16 tokens (one WMMA M tile).
// ---------------------------------------------------------------------------
__global__ __launch_bounds__(256) void fused_head(
        const float* __restrict__ x,
        const int*   __restrict__ idx,
        const float* __restrict__ emb_w,
        const float* __restrict__ gate,
        const float* __restrict__ soft_gate,
        const unsigned short* __restrict__ head_bf,
        const unsigned short* __restrict__ embT_bf,
        float* __restrict__ out,
        float* __restrict__ idx_out) {
    __shared__ unsigned short Pbuf[8][16][72];   // prob tiles, stride-72 kills bank conflicts

    const int tid   = threadIdx.x;
    const int wave  = tid >> 5;
    const int lane  = tid & 31;
    const int row16 = lane & 15;
    const int half  = lane >> 4;
    const int t0    = blockIdx.x * 128 + wave * 16;
    const int tok   = t0 + row16;                 // this lane's A-matrix row / token

    int ve = idx[tok];
    ve = ve < 0 ? 0 : (ve > V - 1 ? V - 1 : ve);

    const float* xrow = x     + (size_t)tok * D;
    const float* erow = emb_w + (size_t)ve  * D;

    // ---- phase 1: g = mean(sigmoid(x * gate)) ; two lanes per token ----
    float gacc = 0.0f;
    #pragma unroll 4
    for (int it = 0; it < 32; ++it) {
        int d = it * 32 + half * 16;
        const float4* xp = (const float4*)(xrow + d);
        const float4* gp = (const float4*)(gate + d);
        #pragma unroll
        for (int q = 0; q < 4; ++q) {
            float4 xv = xp[q], gv = gp[q];
            gacc += sigmoidf(xv.x * gv.x) + sigmoidf(xv.y * gv.y)
                  + sigmoidf(xv.z * gv.z) + sigmoidf(xv.w * gv.w);
        }
    }
    const float g_tok = (gacc + __shfl_xor(gacc, 16, 32)) * (1.0f / 1024.0f);

    // ---- phase 2: logits = Xblend @ head^T, K streamed in chunks of 32 ----
    v8f c0, c1, c2, c3;
    c0 = c1 = c2 = c3 = (v8f){0, 0, 0, 0, 0, 0, 0, 0};
    float sgacc = 0.0f;

    for (int kc = 0; kc < 32; ++kc) {
        const int dbase = kc * 32 + half * 16;    // A layout: half 0 -> K{0..15}
        if (kc < 31) {                             // software prefetch of next x chunk
            __builtin_prefetch(xrow + dbase + 32, 0, 3);
        }
        float xv[16], ev[16], sv[16];
        {
            const float4* xp0 = (const float4*)(xrow + dbase);
            const float4* xp1 = (const float4*)(xrow + dbase + 8);
            const float4* ep0 = (const float4*)(erow + dbase);
            const float4* ep1 = (const float4*)(erow + dbase + 8);
            const float4* sp0 = (const float4*)(soft_gate + dbase);
            const float4* sp1 = (const float4*)(soft_gate + dbase + 8);
            float4 t;
            t = xp0[0]; xv[0]=t.x; xv[1]=t.y; xv[2]=t.z; xv[3]=t.w;
            t = xp0[1]; xv[4]=t.x; xv[5]=t.y; xv[6]=t.z; xv[7]=t.w;
            t = xp1[0]; xv[8]=t.x; xv[9]=t.y; xv[10]=t.z; xv[11]=t.w;
            t = xp1[1]; xv[12]=t.x; xv[13]=t.y; xv[14]=t.z; xv[15]=t.w;
            t = ep0[0]; ev[0]=t.x; ev[1]=t.y; ev[2]=t.z; ev[3]=t.w;
            t = ep0[1]; ev[4]=t.x; ev[5]=t.y; ev[6]=t.z; ev[7]=t.w;
            t = ep1[0]; ev[8]=t.x; ev[9]=t.y; ev[10]=t.z; ev[11]=t.w;
            t = ep1[1]; ev[12]=t.x; ev[13]=t.y; ev[14]=t.z; ev[15]=t.w;
            t = sp0[0]; sv[0]=t.x; sv[1]=t.y; sv[2]=t.z; sv[3]=t.w;
            t = sp0[1]; sv[4]=t.x; sv[5]=t.y; sv[6]=t.z; sv[7]=t.w;
            t = sp1[0]; sv[8]=t.x; sv[9]=t.y; sv[10]=t.z; sv[11]=t.w;
            t = sp1[1]; sv[12]=t.x; sv[13]=t.y; sv[14]=t.z; sv[15]=t.w;
        }
        v8u pk;
        #pragma unroll
        for (int q = 0; q < 8; ++q) {
            float b0 = xv[2*q]   + g_tok * (ev[2*q]   - xv[2*q]);
            float b1 = xv[2*q+1] + g_tok * (ev[2*q+1] - xv[2*q+1]);
            sgacc += sigmoidf(b0 * sv[2*q]) + sigmoidf(b1 * sv[2*q+1]);
            pk[q] = (unsigned)f2bf(b0) | ((unsigned)f2bf(b1) << 16);
        }
        const v16bf a = as16bf(pk);
        // B layout: lane = col, halves split K 0-15 / 16-31, contiguous bf16
        const unsigned short* hb = head_bf + (size_t)row16 * D + kc * 32 + half * 16;
        c0 = wmma_bf16(a, as16bf(*(const v8u*)(hb)),            c0);
        c1 = wmma_bf16(a, as16bf(*(const v8u*)(hb + 16 * D)),   c1);
        c2 = wmma_bf16(a, as16bf(*(const v8u*)(hb + 32 * D)),   c2);
        c3 = wmma_bf16(a, as16bf(*(const v8u*)(hb + 48 * D)),   c3);
    }
    const float sg_tok = (sgacc + __shfl_xor(sgacc, 16, 32)) * (1.0f / 1024.0f);

    // ---- phase 3: softmax over V=64 + argmax; C layout: row = vgpr+8*half, col = nt*16+(lane&15) ----
    v8f p0 = c0, p1 = c1, p2 = c2, p3 = c3;
    #pragma unroll
    for (int j = 0; j < 8; ++j) {
        float l0 = c0[j], l1 = c1[j], l2 = c2[j], l3 = c3[j];
        float m = l0; int mi = row16;
        if (l1 > m) { m = l1; mi = 16 + row16; }
        if (l2 > m) { m = l2; mi = 32 + row16; }
        if (l3 > m) { m = l3; mi = 48 + row16; }
        #pragma unroll
        for (int off = 8; off >= 1; off >>= 1) {   // stays inside 16-lane half
            float om = __shfl_xor(m, off, 32);
            int   oi = __shfl_xor(mi, off, 32);
            if (om > m || (om == m && oi < mi)) { m = om; mi = oi; }
        }
        float e0 = __expf(l0 - m), e1 = __expf(l1 - m);
        float e2 = __expf(l2 - m), e3 = __expf(l3 - m);
        float s = e0 + e1 + e2 + e3;
        #pragma unroll
        for (int off = 8; off >= 1; off >>= 1) s += __shfl_xor(s, off, 32);
        float inv = fast_rcp(s);
        p0[j] = e0 * inv; p1[j] = e1 * inv; p2[j] = e2 * inv; p3[j] = e3 * inv;
        if (row16 == 0) idx_out[t0 + 8 * half + j] = (float)mi;
    }

    // ---- phase 4: C-layout -> A-layout via LDS ----
    #pragma unroll
    for (int j = 0; j < 8; ++j) {
        Pbuf[wave][8 * half + j][row16]      = f2bf(p0[j]);
        Pbuf[wave][8 * half + j][16 + row16] = f2bf(p1[j]);
        Pbuf[wave][8 * half + j][32 + row16] = f2bf(p2[j]);
        Pbuf[wave][8 * half + j][48 + row16] = f2bf(p3[j]);
    }
    __syncthreads();

    v16bf a2[2];
    #pragma unroll
    for (int kk = 0; kk < 2; ++kk) {
        const unsigned short* pr = &Pbuf[wave][row16][0] + kk * 32 + half * 16;
        v4u lo = *(const v4u*)(pr);
        v4u hi = *(const v4u*)(pr + 8);
        a2[kk] = as16bf(__builtin_shufflevector(lo, hi, 0, 1, 2, 3, 4, 5, 6, 7));
    }

    // per-row scalars for the 8 output rows this lane touches
    float gg[8], ss[8]; int vv[8];
    #pragma unroll
    for (int j = 0; j < 8; ++j) {
        int r = 8 * half + j;
        gg[j] = __shfl(g_tok, r, 32);
        ss[j] = __shfl(sg_tok, r, 32);
        vv[j] = __shfl(ve, r, 32);
    }

    // ---- phase 5: soft_emb = P @ emb (K=64), fused final blend + store ----
    for (int nt2 = 0; nt2 < 64; ++nt2) {
        const int dcol = nt2 * 16 + row16;
        v8f acc = (v8f){0, 0, 0, 0, 0, 0, 0, 0};
        const unsigned short* eb = embT_bf + (size_t)dcol * V + half * 16;
        acc = wmma_bf16(a2[0], as16bf(*(const v8u*)(eb)),      acc);
        acc = wmma_bf16(a2[1], as16bf(*(const v8u*)(eb + 32)), acc);
        #pragma unroll
        for (int j = 0; j < 8; ++j) {
            int tr = t0 + 8 * half + j;
            float xv = x[(size_t)tr * D + dcol];
            float evv = emb_w[(size_t)vv[j] * D + dcol];
            float xb = xv + gg[j] * (evv - xv);
            out[(size_t)tr * D + dcol] = xb + ss[j] * (acc[j] - xb);
        }
    }
}

extern "C" void kernel_launch(void* const* d_in, const int* in_sizes, int n_in,
                              void* d_out, int out_size, void* d_ws, size_t ws_size,
                              hipStream_t stream) {
    const float* x        = (const float*)d_in[0];
    const int*   idx      = (const int*)  d_in[1];
    const float* emb_w    = (const float*)d_in[2];
    const float* head_w   = (const float*)d_in[3];
    const float* gate     = (const float*)d_in[4];
    const float* soft_g   = (const float*)d_in[5];

    unsigned short* head_bf = (unsigned short*)d_ws;            // 64*1024 bf16
    unsigned short* embT_bf = head_bf + V * D;                  // 1024*64 bf16

    float* out     = (float*)d_out;
    float* idx_out = out + (size_t)NTOK * D;

    prep_weights<<<(V * D + 255) / 256, 256, 0, stream>>>(head_w, emb_w, head_bf, embT_bf);
    fused_head<<<NTOK / 128, 256, 0, stream>>>(x, idx, emb_w, gate, soft_g,
                                               head_bf, embT_bf, out, idx_out);
}
